// PartitionSpanSegmenter_2319282340677
// MI455X (gfx1250) — matrix operand order repaired and checked by reading, not compiled, not run on hardware
//
#include <hip/hip_runtime.h>
#include <math.h>

#define BB   16
#define TT   512
#define HH   768
#define SS   32
#define LL   50
#define NTY  3
#define HS   3072
#define MTOT (BB*SS*LL)     /* 25600 rows */
#define KCH  768            /* K-chunk staged in LDS (== one repr_ segment) */
#define ROWP (KCH + 8)      /* padded LDS row (bank-conflict avoidance) */
#define NTPW 12             /* N-tiles per wave: (HS/16)/16 */

typedef __attribute__((ext_vector_type(8)))  float  v8f;
typedef __attribute__((ext_vector_type(8)))  __bf16 bf16x8;
typedef __attribute__((ext_vector_type(16))) __bf16 bf16x16;

// ---------------------------------------------------------------------------
// Kernel 0: prefix sums over T (for h_mean), valid_len, zero accumulators.
// ---------------------------------------------------------------------------
__global__ void k_prepare(const float* __restrict__ h,
                          const int*   __restrict__ amask,
                          float* __restrict__ pre,
                          int*   __restrict__ vlen,
                          float* __restrict__ accum)
{
    int idx = blockIdx.x * blockDim.x + threadIdx.x;
    if (idx < 3) accum[idx] = 0.f;
    if (idx < BB) {
        int s = 0;
        for (int t = 0; t < TT; ++t) s += amask[idx * TT + t];
        vlen[idx] = s;
    }
    if (idx < BB * HH) {
        int b = idx / HH, hh = idx % HH;
        const float* hp = h   + (size_t)b * TT * HH + hh;
        float*       pp = pre + (size_t)b * (TT + 1) * HH + hh;
        float run = 0.f;
        pp[0] = 0.f;
        for (int t = 0; t < TT; ++t) {
            run += hp[(size_t)t * HH];
            pp[(size_t)(t + 1) * HH] = run;
        }
    }
}

// ---------------------------------------------------------------------------
// Kernel 1: split W1 (fp32, [K][N]) into bf16 hi/lo, transposed to [N][K]
// so B fragments are contiguous 32B loads per lane.
// ---------------------------------------------------------------------------
__global__ void k_splitW1(const float* __restrict__ W1,
                          __bf16* __restrict__ Whi,
                          __bf16* __restrict__ Wlo)
{
    size_t idx = (size_t)blockIdx.x * blockDim.x + threadIdx.x;
    if (idx >= (size_t)HS * HS) return;
    int k = (int)(idx / HS);
    int n = (int)(idx % HS);
    float v = W1[idx];
    __bf16 hi = (__bf16)v;
    __bf16 lo = (__bf16)(v - (float)hi);
    size_t o = (size_t)n * HS + k;
    Whi[o] = hi;
    Wlo[o] = lo;
}

// ---------------------------------------------------------------------------
// Kernel 2: fused GEMM (split-bf16 WMMA, fp32 accum) + bias + exact GELU +
// projection onto w_score / W_type. Block = 16 candidate rows, 16 waves.
// Each wave owns a contiguous 192-column slab so all 12 N-tile B loads fold
// into 24-bit immediate offsets (stride 16*HS*2 = 98304 B per tile).
// ---------------------------------------------------------------------------
__global__ __launch_bounds__(512) void k_gemm(
    const float*  __restrict__ h,
    const float*  __restrict__ pre,
    const __bf16* __restrict__ Whi,
    const __bf16* __restrict__ Wlo,
    const int*    __restrict__ starts,
    const float*  __restrict__ b1,
    const float*  __restrict__ w_score,
    const float*  __restrict__ W_type,
    float* __restrict__ scores,
    float* __restrict__ tlog)
{
    __shared__ __align__(16) __bf16 Ahi[16 * ROWP];
    __shared__ __align__(16) __bf16 Alo[16 * ROWP];
    __shared__ float red[16][16][4];
    __shared__ int rb[16], rsc[16], rpc[16], rl[16];

    const int tid = threadIdx.x;
    const int m0  = blockIdx.x * 16;

    if (tid < 16) {
        int m   = m0 + tid;
        int b   = m / (SS * LL);
        int rem = m % (SS * LL);
        int si  = rem / LL;
        int l   = rem % LL;
        int s   = starts[b * SS + si];
        int p   = s + l;
        rb[tid]  = b;
        rsc[tid] = min(max(s, 0), TT - 1);
        rpc[tid] = min(max(p, 0), TT - 1);
        rl[tid]  = l;
    }
    __syncthreads();

    const int wave = tid >> 5;
    const int lane = tid & 31;
    const int g    = lane >> 4;   // half-wave: selects K sub-range of frags
    const int ln   = lane & 15;   // M for A-frag, N for B/C frags

    v8f zf = {};
    v8f acc[NTPW];
#pragma unroll
    for (int i = 0; i < NTPW; ++i) acc[i] = zf;

    // Per-lane base for its slab: column (wave*192 + ln), K advances inside.
    const size_t laneBase = (size_t)(wave * 192 + ln) * HS;

    for (int pass = 0; pass < 4; ++pass) {
        __syncthreads();
        // Build A chunk in LDS: columns [pass*768, pass*768+768) of repr_.
        for (int idx = tid; idx < 16 * KCH; idx += 512) {
            int r  = idx / KCH;
            int cc = idx - r * KCH;           // feature index within segment
            int b  = rb[r];
            float v;
            if (pass == 0) {                  // h_start
                v = h[((size_t)b * TT + rsc[r]) * HH + cc];
            } else if (pass == 1) {           // h_end
                v = h[((size_t)b * TT + rpc[r]) * HH + cc];
            } else if (pass == 2) {           // h_mean via prefix sums
                const float* pb = pre + (size_t)b * (TT + 1) * HH;
                v = (pb[(size_t)(rpc[r] + 1) * HH + cc] -
                     pb[(size_t)rsc[r] * HH + cc]) / (float)(rl[r] + 1);
            } else {                          // cls
                v = h[(size_t)b * TT * HH + cc];
            }
            __bf16 hi = (__bf16)v;
            __bf16 lo = (__bf16)(v - (float)hi);
            Ahi[r * ROWP + cc] = hi;
            Alo[r * ROWP + cc] = lo;
        }
        __syncthreads();

        const int kbase = pass * KCH;
        for (int step = 0; step < KCH / 32; ++step) {
            const int k0 = step * 32;
            // A fragment (16x32 bf16 layout): lane ln holds row M=ln,
            // K = k0 + g*16 + [0..15].
            const __bf16* arh = Ahi + ln * ROWP + k0 + g * 16;
            const __bf16* arl = Alo + ln * ROWP + k0 + g * 16;
            bf16x8 a0h = *(const bf16x8*)(arh);
            bf16x8 a1h = *(const bf16x8*)(arh + 8);
            bf16x8 a0l = *(const bf16x8*)(arl);
            bf16x8 a1l = *(const bf16x8*)(arl + 8);
            bf16x16 ah = __builtin_shufflevector(a0h, a1h,
                0,1,2,3,4,5,6,7,8,9,10,11,12,13,14,15);
            bf16x16 al = __builtin_shufflevector(a0l, a1l,
                0,1,2,3,4,5,6,7,8,9,10,11,12,13,14,15);

            // B fragment (32x16 bf16 layout): lane ln holds column N,
            // K = k0 + g*16 + [0..15]  -> contiguous in transposed W1.
            const size_t kofs = laneBase + (size_t)(kbase + k0 + g * 16);
            const __bf16* bpH = Whi + kofs;
            const __bf16* bpL = Wlo + kofs;
#pragma unroll
            for (int nt = 0; nt < NTPW; ++nt) {
                // per-tile delta = 16*HS elems = 98304 B: fits imm offset
                bf16x16 bh = *(const bf16x16*)(bpH + nt * (16 * HS));
                bf16x16 bl = *(const bf16x16*)(bpL + nt * (16 * HS));
                // split-bf16: ah*bh + ah*bl + al*bh  (~fp32 accuracy)
                acc[nt] = __builtin_amdgcn_wmma_f32_16x16x32_bf16(
                    false, ah, false, bh, (short)0, acc[nt], false, false);
                acc[nt] = __builtin_amdgcn_wmma_f32_16x16x32_bf16(
                    false, ah, false, bl, (short)0, acc[nt], false, false);
                acc[nt] = __builtin_amdgcn_wmma_f32_16x16x32_bf16(
                    false, al, false, bh, (short)0, acc[nt], false, false);
            }
        }
    }

    // Fused epilogue: bias + exact GELU, project onto w_score / W_type.
    float sacc[8], t0a[8], t1a[8], t2a[8];
#pragma unroll
    for (int i = 0; i < 8; ++i) { sacc[i] = 0.f; t0a[i] = 0.f; t1a[i] = 0.f; t2a[i] = 0.f; }

#pragma unroll
    for (int nt = 0; nt < NTPW; ++nt) {
        int n = wave * 192 + nt * 16 + ln;
        float bias = b1[n];
        float wsc  = w_score[n];
        float w0   = W_type[n * NTY + 0];
        float w1v  = W_type[n * NTY + 1];
        float w2   = W_type[n * NTY + 2];
#pragma unroll
        for (int i = 0; i < 8; ++i) {
            float x  = acc[nt][i] + bias;
            float ge = 0.5f * x * (1.f + erff(x * 0.70710678118654752440f));
            sacc[i] += ge * wsc;
            t0a[i]  += ge * w0;
            t1a[i]  += ge * w1v;
            t2a[i]  += ge * w2;
        }
    }

    // Reduce over the 16 lanes of each half (C-layout: half g owns M=g*8+i).
#pragma unroll
    for (int off = 1; off < 16; off <<= 1) {
#pragma unroll
        for (int i = 0; i < 8; ++i) {
            sacc[i] += __shfl_xor(sacc[i], off, 32);
            t0a[i]  += __shfl_xor(t0a[i],  off, 32);
            t1a[i]  += __shfl_xor(t1a[i],  off, 32);
            t2a[i]  += __shfl_xor(t2a[i],  off, 32);
        }
    }
    if (ln == 0) {
#pragma unroll
        for (int i = 0; i < 8; ++i) {
            int r = g * 8 + i;
            red[wave][r][0] = sacc[i];
            red[wave][r][1] = t0a[i];
            red[wave][r][2] = t1a[i];
            red[wave][r][3] = t2a[i];
        }
    }
    __syncthreads();
    if (tid < 64) {
        int r = tid >> 2, ch = tid & 3;
        float sum = 0.f;
#pragma unroll
        for (int w = 0; w < 16; ++w) sum += red[w][r][ch];
        int m = m0 + r;
        if (ch == 0) scores[m] = sum;
        else         tlog[(size_t)m * NTY + (ch - 1)] = sum;
    }
}

// ---------------------------------------------------------------------------
// Kernel 3: per-(b,s) masked log-softmax over L, gold gathers, CE, masked sums.
// ---------------------------------------------------------------------------
__global__ void k_loss(const float* __restrict__ scores,
                       const float* __restrict__ tlog,
                       const int*   __restrict__ starts,
                       const int*   __restrict__ ends,
                       const int*   __restrict__ types,
                       const int*   __restrict__ vlen,
                       const float* __restrict__ b_score,
                       const float* __restrict__ b_type,
                       float* __restrict__ accum)
{
    int idx = blockIdx.x * blockDim.x + threadIdx.x;
    if (idx >= BB * SS) return;
    int b  = idx / SS;
    int lt = max(1, vlen[b] - 2);
    int s  = starts[idx];
    int e  = ends[idx];
    int gidx = e - s;
    bool valid = (s >= 1) && (s <= lt) && (e >= s) && (e <= lt) && (gidx < LL);
    int emax = min(s + LL - 1, lt);
    float bs = b_score[0];
    const float* sp = scores + (size_t)idx * LL;

    float mx = -1e30f;
    for (int l = 0; l < LL; ++l) {
        float v = (s + l <= emax) ? (sp[l] + bs) : -1.0e9f;
        mx = fmaxf(mx, v);
    }
    float sum = 0.f;
    for (int l = 0; l < LL; ++l) {
        float v = (s + l <= emax) ? (sp[l] + bs) : -1.0e9f;
        sum += expf(v - mx);
    }
    float logZ = mx + logf(sum);
    int gi = min(max(gidx, 0), LL - 1);
    float vg = (s + gi <= emax) ? (sp[gi] + bs) : -1.0e9f;
    float end_loss = logZ - vg;

    const float* tp = tlog + (size_t)(idx * LL + gi) * NTY;
    float t0 = tp[0] + b_type[0];
    float t1 = tp[1] + b_type[1];
    float t2 = tp[2] + b_type[2];
    float tm = fmaxf(t0, fmaxf(t1, t2));
    float tz = tm + logf(expf(t0 - tm) + expf(t1 - tm) + expf(t2 - tm));
    int tg = min(max(types[idx], 0), NTY - 1);
    float tv = (tg == 0) ? t0 : ((tg == 1) ? t1 : t2);
    float type_loss = tz - tv;

    if (valid) {
        atomicAdd(accum + 0, end_loss);
        atomicAdd(accum + 1, type_loss);
        atomicAdd(accum + 2, 1.0f);
    }
}

__global__ void k_final(const float* __restrict__ accum, float* __restrict__ out)
{
    float n = accum[2];
    float denom = fmaxf(n, 1.f);
    out[0] = (n > 0.f) ? (accum[0] + accum[1]) / denom : 0.f;
}

// ---------------------------------------------------------------------------
extern "C" void kernel_launch(void* const* d_in, const int* in_sizes, int n_in,
                              void* d_out, int out_size, void* d_ws, size_t ws_size,
                              hipStream_t stream)
{
    const float* h      = (const float*)d_in[0];
    const int*   amask  = (const int*)  d_in[1];
    const int*   starts = (const int*)  d_in[2];
    const int*   ends   = (const int*)  d_in[3];
    const int*   types  = (const int*)  d_in[4];
    const float* W1     = (const float*)d_in[5];
    const float* b1     = (const float*)d_in[6];
    const float* wsc    = (const float*)d_in[7];
    const float* bsc    = (const float*)d_in[8];
    const float* Wty    = (const float*)d_in[9];
    const float* bty    = (const float*)d_in[10];
    float* out = (float*)d_out;

    char* ws = (char*)d_ws;
    size_t off = 0;
    float*  pre    = (float*) (ws + off); off += (size_t)BB * (TT + 1) * HH * 4;
    __bf16* Whi    = (__bf16*)(ws + off); off += (size_t)HS * HS * 2;
    __bf16* Wlo    = (__bf16*)(ws + off); off += (size_t)HS * HS * 2;
    float*  scoreA = (float*) (ws + off); off += (size_t)MTOT * 4;
    float*  tlogA  = (float*) (ws + off); off += (size_t)MTOT * NTY * 4;
    int*    vlen   = (int*)   (ws + off); off += 64;
    float*  accum  = (float*) (ws + off); off += 64;

    k_prepare<<<(BB * HH + 255) / 256, 256, 0, stream>>>(h, amask, pre, vlen, accum);
    k_splitW1<<<(int)(((size_t)HS * HS + 255) / 256), 256, 0, stream>>>(W1, Whi, Wlo);
    k_gemm<<<MTOT / 16, 512, 0, stream>>>(h, pre, Whi, Wlo, starts, b1, wsc, Wty,
                                          scoreA, tlogA);
    k_loss<<<(BB * SS + 255) / 256, 256, 0, stream>>>(scoreA, tlogA, starts, ends,
                                                      types, vlen, bsc, bty, accum);
    k_final<<<1, 1, 0, stream>>>(accum, out);
}
